// SPDTransform1_38620345925760
// MI455X (gfx1250) — matrix operand order repaired and checked by reading, not compiled, not run on hardware
//
#include <hip/hip_runtime.h>
#include <hip/hip_bf16.h>

typedef __bf16 bf16;
typedef __attribute__((ext_vector_type(8)))  __bf16 v8bf;
typedef __attribute__((ext_vector_type(16))) __bf16 v16bf;
typedef __attribute__((ext_vector_type(8)))  float  v8f;

#define WMMA_BF16(A, Bv, C) \
  __builtin_amdgcn_wmma_f32_16x16x32_bf16(false, (A), false, (Bv), (short)0, (C), false, false)

__device__ __forceinline__ v8f vzero8() {
  v8f z;
#pragma unroll
  for (int i = 0; i < 8; ++i) z[i] = 0.0f;
  return z;
}

// Load 16 bf16 as two contiguous 16B chunks (lo = elements 0..7, hi = 8..15).
__device__ __forceinline__ v16bf ld16(const bf16* plo, const bf16* phi) {
  v8bf lo = *reinterpret_cast<const v8bf*>(plo);
  v8bf hi = *reinterpret_cast<const v8bf*>(phi);
  return __builtin_shufflevector(lo, hi, 0, 1, 2, 3, 4, 5, 6, 7, 8, 9, 10, 11, 12, 13, 14, 15);
}

// ---------------------------------------------------------------------------
// CDNA5 async global->LDS copy (ASYNCcnt-tracked, bypasses VGPRs).
// GVS addressing: mem = SGPR_base + VGPR_offset; LDS dest = VGPR lds_off.
// ---------------------------------------------------------------------------
__device__ __forceinline__ void async_copy_b128(unsigned lds_off, const bf16* gbase,
                                                unsigned voff_bytes) {
  asm volatile("global_load_async_to_lds_b128 %0, %1, %2"
               :
               : "v"(lds_off), "v"(voff_bytes),
                 "s"((unsigned long long)(uintptr_t)gbase)
               : "memory");
}

__device__ __forceinline__ void wait_async0() {
  asm volatile("s_wait_asynccnt 0x0" ::: "memory");
}

__device__ __forceinline__ unsigned lds_addr(const void* p) {
  // Generic (flat) pointers to LDS keep the LDS byte offset in the low 32 bits.
  return (unsigned)(uintptr_t)p;
}

// ---------------------------------------------------------------------------
// Pass A: conversion kernels
// ---------------------------------------------------------------------------

// dst[cols][rows] = (bf16) src[rows][cols]   (tiled transpose-convert)
__global__ __launch_bounds__(256) void transpose_cvt_bf16(const float* __restrict__ src,
                                                          bf16* __restrict__ dst,
                                                          int rows, int cols) {
  __shared__ float tile[32][33];
  const int x = blockIdx.x * 32 + threadIdx.x;   // src col
  const int y0 = blockIdx.y * 32;                // src row base
#pragma unroll
  for (int j = threadIdx.y; j < 32; j += 8)
    tile[j][threadIdx.x] = src[(size_t)(y0 + j) * cols + x];
  __syncthreads();
  const int xo = y0 + threadIdx.x;               // dst col (= src row)
  const int yo0 = blockIdx.x * 32;               // dst row base (= src col)
#pragma unroll
  for (int j = threadIdx.y; j < 32; j += 8)
    dst[(size_t)(yo0 + j) * rows + xo] = (bf16)tile[threadIdx.x][j];
}

__global__ __launch_bounds__(256) void cvt_bf16(const float* __restrict__ src,
                                                bf16* __restrict__ dst, int n) {
  int i = blockIdx.x * blockDim.x + threadIdx.x;
  if (i < n) dst[i] = (bf16)src[i];
}

// ---------------------------------------------------------------------------
// Pass B: h = silu(Y @ Y_W1 + b1)     Y:[4096][64] f32, W1T:[4096 n][64 k] bf16
// ---------------------------------------------------------------------------
__global__ __launch_bounds__(256) void gemm1_silu(const float* __restrict__ Y,
                                                  const bf16* __restrict__ W1T,
                                                  const float* __restrict__ b1,
                                                  bf16* __restrict__ H) {
  const int l = threadIdx.x & 31, w = threadIdx.x >> 5;
  const int r0 = blockIdx.x * 128 + w * 16;
  const int c0 = blockIdx.y * 64;
  const int hs8 = (l >= 16) ? 8 : 0;
  const int hs16 = (l >= 16) ? 16 : 0;
  const int moff = (l >= 16) ? 8 : 0;

  v8f acc[4];
#pragma unroll
  for (int nt = 0; nt < 4; ++nt) acc[nt] = vzero8();

  const int m = r0 + (l & 15);
  const float* yrow = Y + (size_t)m * 64;

#pragma unroll
  for (int ks = 0; ks < 64; ks += 32) {
    v16bf a;
#pragma unroll
    for (int i = 0; i < 8; ++i) a[i] = (bf16)yrow[ks + hs8 + i];
#pragma unroll
    for (int i = 0; i < 8; ++i) a[i + 8] = (bf16)yrow[ks + 16 + hs8 + i];
#pragma unroll
    for (int nt = 0; nt < 4; ++nt) {
      const int n = c0 + nt * 16 + (l & 15);
      const bf16* p = W1T + (size_t)n * 64 + ks + hs8;
      v16bf b = ld16(p, p + 16);
      acc[nt] = WMMA_BF16(a, b, acc[nt]);
    }
  }

#pragma unroll
  for (int nt = 0; nt < 4; ++nt) {
    const int col = c0 + nt * 16 + (l & 15);
    const float bb = b1[col];
#pragma unroll
    for (int v = 0; v < 8; ++v) {
      const int row = r0 + v + moff;
      const float x = acc[nt][v] + bb;
      const float s = x / (1.0f + __expf(-x));
      H[(size_t)row * 4096 + col] = (bf16)s;
    }
  }
}

// ---------------------------------------------------------------------------
// Pass C: emb = H @ Y_W2 + b2     H:[4096][4096] bf16, W2T:[4096 n][4096 k] bf16
// 128x128 block tile, BK=32, double-buffered LDS filled with CDNA5 async
// global->LDS copies (overlap next tile's loads with current tile's WMMAs).
// 8 waves as 4x2; each wave: 2 M-tiles x 4 N-tiles accumulators.
// ---------------------------------------------------------------------------
__global__ __launch_bounds__(256) void gemm2_bias(const bf16* __restrict__ A,
                                                  const bf16* __restrict__ BT,
                                                  const float* __restrict__ bias,
                                                  bf16* __restrict__ C) {
  __shared__ bf16 As[2][128 * 40];   // BK=32, row stride 40 (80B, 16B-aligned)
  __shared__ bf16 Bs[2][128 * 40];
  const int tid = threadIdx.x;
  const int l = tid & 31, w = tid >> 5;
  const int m0 = blockIdx.x * 128, n0 = blockIdx.y * 128;
  const int wm = w >> 1, wn = w & 1;
  const int hs8 = (l >= 16) ? 8 : 0;
  const int hs16 = (l >= 16) ? 16 : 0;
  const int moff = (l >= 16) ? 8 : 0;

  const unsigned ldsA0 = lds_addr(&As[0][0]), ldsA1 = lds_addr(&As[1][0]);
  const unsigned ldsB0 = lds_addr(&Bs[0][0]), ldsB1 = lds_addr(&Bs[1][0]);

  // 512 16B chunks per matrix per tile, 2 per thread.
  const int ca = tid, cb = tid + 256;
  const int ra = ca >> 2, ka = (ca & 3) * 8;
  const int rb = cb >> 2, kb = (cb & 3) * 8;

  v8f acc[2][4];
#pragma unroll
  for (int i = 0; i < 2; ++i)
#pragma unroll
    for (int j = 0; j < 4; ++j) acc[i][j] = vzero8();

  auto issue_tiles = [&](unsigned la, unsigned lb, int k0) {
    const bf16* aB = A + (size_t)m0 * 4096 + k0;
    const bf16* bB = BT + (size_t)n0 * 4096 + k0;
    async_copy_b128(la + (unsigned)(ra * 40 + ka) * 2, aB, (unsigned)(ra * 4096 + ka) * 2);
    async_copy_b128(la + (unsigned)(rb * 40 + kb) * 2, aB, (unsigned)(rb * 4096 + kb) * 2);
    async_copy_b128(lb + (unsigned)(ra * 40 + ka) * 2, bB, (unsigned)(ra * 4096 + ka) * 2);
    async_copy_b128(lb + (unsigned)(rb * 40 + kb) * 2, bB, (unsigned)(rb * 4096 + kb) * 2);
  };

  issue_tiles(ldsA0, ldsB0, 0);

  for (int k0 = 0; k0 < 4096; k0 += 32) {
    const int cur = (k0 >> 5) & 1;
    wait_async0();        // current buffer's async copies have landed in LDS
    __syncthreads();      // ... for every wave in the group
    if (k0 + 32 < 4096)   // prefetch next K-tile while we compute
      issue_tiles(cur ? ldsA0 : ldsA1, cur ? ldsB0 : ldsB1, k0 + 32);

    const bf16* Ab = As[cur];
    const bf16* Bb = Bs[cur];
    v16bf afrag[2];
#pragma unroll
    for (int mt = 0; mt < 2; ++mt) {
      const int m = wm * 32 + mt * 16 + (l & 15);
      const bf16* p = Ab + m * 40 + hs8;
      afrag[mt] = ld16(p, p + 16);
    }
#pragma unroll
    for (int nt = 0; nt < 4; ++nt) {
      const int n = wn * 64 + nt * 16 + (l & 15);
      const bf16* p = Bb + n * 40 + hs8;
      v16bf bfrag = ld16(p, p + 16);
      acc[0][nt] = WMMA_BF16(afrag[0], bfrag, acc[0][nt]);
      acc[1][nt] = WMMA_BF16(afrag[1], bfrag, acc[1][nt]);
    }
    __syncthreads();      // all reads of 'cur' done before it is refilled
  }

#pragma unroll
  for (int mt = 0; mt < 2; ++mt)
#pragma unroll
    for (int nt = 0; nt < 4; ++nt) {
      const int col = n0 + wn * 64 + nt * 16 + (l & 15);
      const float bv = bias[col];
#pragma unroll
      for (int v = 0; v < 8; ++v) {
        const int row = m0 + wm * 32 + mt * 16 + v + moff;
        C[(size_t)row * 4096 + col] = (bf16)(acc[mt][nt][v] + bv);
      }
    }
}

// ---------------------------------------------------------------------------
// Pass D: per-sample  out = E (W^T xp W) E^T  chained in LDS.
// ---------------------------------------------------------------------------
__device__ __forceinline__ void gemm64_2tile(const bf16* Als, const bf16* BTls,
                                             int mt, int nt0, int l, int hs8, int hs16,
                                             v8f& acc0, v8f& acc1) {
#pragma unroll
  for (int ks = 0; ks < 64; ks += 32) {
    const int m = mt * 16 + (l & 15);
    v16bf a = ld16(Als + m * 80 + ks + hs8, Als + m * 80 + ks + 16 + hs8);
    const int n0 = nt0 * 16 + (l & 15);
    v16bf b0 = ld16(BTls + n0 * 80 + ks + hs8, BTls + n0 * 80 + ks + 16 + hs8);
    v16bf b1 = ld16(BTls + (n0 + 16) * 80 + ks + hs8, BTls + (n0 + 16) * 80 + ks + 16 + hs8);
    acc0 = WMMA_BF16(a, b0, acc0);
    acc1 = WMMA_BF16(a, b1, acc1);
  }
}

__device__ __forceinline__ void store_tr(bf16* Outls, int mt, int nt, int l, int moff,
                                         const v8f& acc) {
  const int n = nt * 16 + (l & 15);
#pragma unroll
  for (int v = 0; v < 8; ++v) Outls[n * 80 + mt * 16 + v + moff] = (bf16)acc[v];
}

__device__ __forceinline__ void store_rm(bf16* Outls, int mt, int nt, int l, int moff,
                                         const v8f& acc) {
  const int n = nt * 16 + (l & 15);
#pragma unroll
  for (int v = 0; v < 8; ++v) Outls[(mt * 16 + v + moff) * 80 + n] = (bf16)acc[v];
}

__global__ __launch_bounds__(256) void spd_congruence(const float* __restrict__ X,
                                                      const bf16* __restrict__ Wb,
                                                      const bf16* __restrict__ Ebf,
                                                      float* __restrict__ Out) {
  __shared__ bf16 Sxp[64 * 80];  // xp, later M1^T
  __shared__ bf16 SwT[64 * 80];  // W^T  (SwT[n][k] = W[k][n])
  __shared__ bf16 Se[64 * 80];   // emb_Y row-major
  __shared__ bf16 St[64 * 80];   // T1^T, later T2 row-major
  const int b = blockIdx.x;
  const int tid = threadIdx.x, l = tid & 31, w = tid >> 5;
  const int hs8 = (l >= 16) ? 8 : 0;
  const int hs16 = (l >= 16) ? 16 : 0;
  const int moff = (l >= 16) ? 8 : 0;

  // emb_Y row-major: pure copy -> async global->LDS (512 16B chunks, 2/thread).
  {
    const unsigned ldsSe = lds_addr(&Se[0]);
    const bf16* eB = Ebf + (size_t)b * 4096;
#pragma unroll
    for (int i = 0; i < 2; ++i) {
      const int c = tid + i * 256;
      const int r = c >> 3, kc = (c & 7) * 8;
      async_copy_b128(ldsSe + (unsigned)(r * 80 + kc) * 2, eB, (unsigned)(r * 64 + kc) * 2);
    }
  }
  // xp (SPD embed + pad diag) and W^T need compute/transpose: direct fill.
  for (int idx = tid; idx < 4096; idx += 256) {
    const int r = idx >> 6, c = idx & 63;
    const float xv = (r < 48 && c < 48) ? X[(size_t)b * 2304 + r * 48 + c]
                                        : ((r == c) ? 1.0f : 0.0f);
    Sxp[r * 80 + c] = (bf16)xv;
    SwT[r * 80 + c] = Wb[c * 64 + r];
  }
  wait_async0();
  __syncthreads();

  const int mt = w >> 1;
  const int nt0 = (2 * w) & 3;
  v8f a0, a1;

  // S1: T1 = xp @ W      (A=Sxp, BT=SwT)  -> St = T1^T
  a0 = vzero8(); a1 = vzero8();
  gemm64_2tile(Sxp, SwT, mt, nt0, l, hs8, hs16, a0, a1);
  store_tr(St, mt, nt0, l, moff, a0);
  store_tr(St, mt, nt0 + 1, l, moff, a1);
  __syncthreads();

  // S2: M1 = W^T @ T1    (A=SwT, BT=St)   -> Sxp = M1^T
  a0 = vzero8(); a1 = vzero8();
  gemm64_2tile(SwT, St, mt, nt0, l, hs8, hs16, a0, a1);
  store_tr(Sxp, mt, nt0, l, moff, a0);
  store_tr(Sxp, mt, nt0 + 1, l, moff, a1);
  __syncthreads();

  // S3: T2 = E @ M1      (A=Se, BT=Sxp)   -> St = T2 row-major
  a0 = vzero8(); a1 = vzero8();
  gemm64_2tile(Se, Sxp, mt, nt0, l, hs8, hs16, a0, a1);
  store_rm(St, mt, nt0, l, moff, a0);
  store_rm(St, mt, nt0 + 1, l, moff, a1);
  __syncthreads();

  // S4: out = T2 @ E^T   (A=St, BT=Se; (E^T)[k][n] = E[n][k] = Se rows)
  a0 = vzero8(); a1 = vzero8();
  gemm64_2tile(St, Se, mt, nt0, l, hs8, hs16, a0, a1);

  float* outb = Out + (size_t)b * 4096;
#pragma unroll
  for (int v = 0; v < 8; ++v) {
    const int m = mt * 16 + v + moff;
    outb[m * 64 + nt0 * 16 + (l & 15)] = a0[v];
    outb[m * 64 + (nt0 + 1) * 16 + (l & 15)] = a1[v];
  }
}

// ---------------------------------------------------------------------------
// Launch
// ---------------------------------------------------------------------------
extern "C" void kernel_launch(void* const* d_in, const int* in_sizes, int n_in,
                              void* d_out, int out_size, void* d_ws, size_t ws_size,
                              hipStream_t stream) {
  (void)in_sizes; (void)n_in; (void)out_size; (void)ws_size;
  const float* X = (const float*)d_in[0];
  // d_in[1] = t  (unused by the reference)
  const float* Y = (const float*)d_in[2];
  const float* W = (const float*)d_in[3];
  const float* YW1 = (const float*)d_in[4];
  const float* Yb1 = (const float*)d_in[5];
  const float* YW2 = (const float*)d_in[6];
  const float* Yb2 = (const float*)d_in[7];
  float* Out = (float*)d_out;

  bf16* ws = (bf16*)d_ws;
  bf16* W1T = ws;                            // [4096][64]
  bf16* W2T = W1T + (size_t)4096 * 64;       // [4096][4096]  (Y_W2 transposed)
  bf16* Wb = W2T + (size_t)4096 * 4096;      // [64][64]
  bf16* H = Wb + (size_t)64 * 64;            // [4096][4096]
  bf16* E = H + (size_t)4096 * 4096;         // [4096][4096]

  transpose_cvt_bf16<<<dim3(128, 2), dim3(32, 8), 0, stream>>>(YW1, W1T, 64, 4096);
  transpose_cvt_bf16<<<dim3(128, 128), dim3(32, 8), 0, stream>>>(YW2, W2T, 4096, 4096);
  cvt_bf16<<<dim3(16), dim3(256), 0, stream>>>(W, Wb, 64 * 64);
  gemm1_silu<<<dim3(32, 64), dim3(256), 0, stream>>>(Y, W1T, Yb1, H);
  gemm2_bias<<<dim3(32, 32), dim3(256), 0, stream>>>(H, W2T, Yb2, E);
  spd_congruence<<<dim3(4096), dim3(256), 0, stream>>>(X, Wb, E, Out);
}
